// VectorQuantizer_79731772883123
// MI455X (gfx1250) — matrix operand order repaired and checked by reading, not compiled, not run on hardware
//
#include <hip/hip_runtime.h>
#include <hip/hip_bf16.h>
#include <math.h>

typedef __attribute__((ext_vector_type(16))) _Float16 v16h;
typedef __attribute__((ext_vector_type(8)))  _Float16 v8h;
typedef __attribute__((ext_vector_type(8)))  float    v8f;
typedef __attribute__((ext_vector_type(4)))  int      v4i;

constexpr int  Bb = 8;
constexpr int  D  = 256;
constexpr int  S  = 16 * 32 * 32;        // 16384 spatial positions per batch
constexpr int  N  = Bb * S;              // 131072 rows
constexpr int  K  = 512;                 // codebook entries
constexpr long long ND = (long long)N * D;   // 33554432

#define AS1 __attribute__((address_space(1)))
#define AS3 __attribute__((address_space(3)))

#if defined(__has_builtin)
#  if __has_builtin(__builtin_amdgcn_global_load_async_to_lds_b128)
#    define HAVE_ASYNC_LDS_BUILTIN 1
#  endif
#  if __has_builtin(__builtin_amdgcn_s_wait_asynccnt)
#    define HAVE_WAIT_ASYNC_BUILTIN 1
#  endif
#endif

// Async copy of 16 bytes global -> LDS (per-lane), tracked by ASYNCcnt.
__device__ __forceinline__ void async_b128(const void* g, void* l) {
#if defined(HAVE_ASYNC_LDS_BUILTIN)
    __builtin_amdgcn_global_load_async_to_lds_b128(
        (AS1 v4i*)(AS1 void*)g, (AS3 v4i*)(AS3 void*)l, 0, 0);
#else
    unsigned lds_off = (unsigned)(uintptr_t)(AS3 void*)l;
    asm volatile("global_load_async_to_lds_b128 %0, %1, off"
                 :: "v"(lds_off), "v"(g) : "memory");
#endif
}

__device__ __forceinline__ void wait_async0() {
#if defined(HAVE_WAIT_ASYNC_BUILTIN)
    __builtin_amdgcn_s_wait_asynccnt(0);
#else
    asm volatile("s_wait_asynccnt 0x0" ::: "memory");
#endif
}

// ---------------------------------------------------------------------------
// Zero histogram + loss accumulator (must run every call: graph replays reuse ws)
// ---------------------------------------------------------------------------
__global__ void vq_zero(int* __restrict__ hist, float* __restrict__ loss) {
    int i = blockIdx.x * 256 + threadIdx.x;
    if (i < K) hist[i] = 0;
    if (i == 0) *loss = 0.0f;
}

// ---------------------------------------------------------------------------
// Codebook prep: f32 -> f16 copy + per-row squared norms.  One block per code.
// ---------------------------------------------------------------------------
__global__ __launch_bounds__(256) void vq_prep_e(const float* __restrict__ e,
                                                 _Float16* __restrict__ e16,
                                                 float* __restrict__ enorm) {
    __shared__ float red[256];
    int k = blockIdx.x;
    int t = threadIdx.x;
    float v = e[k * D + t];
    e16[k * D + t] = (_Float16)v;
    red[t] = v * v;
    __syncthreads();
    for (int st = 128; st > 0; st >>= 1) {
        if (t < st) red[t] += red[t + st];
        __syncthreads();
    }
    if (t == 0) enorm[k] = red[0];
}

// ---------------------------------------------------------------------------
// Transpose+convert: z[B,D,S] f32  ->  zt[N=B*S, D] f16   (32x32 LDS tiles)
// ---------------------------------------------------------------------------
__global__ __launch_bounds__(256) void vq_transpose(const float* __restrict__ z,
                                                    _Float16* __restrict__ zt) {
    __shared__ float tile[32][33];
    int b  = blockIdx.z;
    int ts = blockIdx.x * 32;   // spatial tile base
    int td = blockIdx.y * 32;   // dim tile base
    int tx = threadIdx.x;       // 0..31
    int ty = threadIdx.y;       // 0..7

    const float* zb = z + (size_t)b * D * S;
#pragma unroll
    for (int j = 0; j < 4; ++j) {
        int d = td + ty + j * 8;
        tile[ty + j * 8][tx] = zb[(size_t)d * S + (ts + tx)];
    }
    __syncthreads();
#pragma unroll
    for (int j = 0; j < 4; ++j) {
        int s = ts + ty + j * 8;
        int d = td + tx;
        zt[((size_t)b * S + s) * D + d] = (_Float16)tile[tx][ty + j * 8];
    }
}

// ---------------------------------------------------------------------------
// WMMA distance GEMM + argmin.
// Block = 256 threads = 8 waves; each wave owns a 16-row M tile.
// Codebook tiles (16 codes x 256 dims, 8 KB) are double-buffered in LDS via
// async global->LDS copies; prefetch of tile kt+1 overlaps WMMA on tile kt.
// score[n][k] = ||e_k||^2 - 2 * <z_n, e_k>   (|z_n|^2 constant per row -> dropped)
// ---------------------------------------------------------------------------
__global__ __launch_bounds__(256) void vq_argmin(const _Float16* __restrict__ zt,
                                                 const _Float16* __restrict__ e16,
                                                 const float* __restrict__ enorm,
                                                 int* __restrict__ idx_i,
                                                 float* __restrict__ idx_f,
                                                 int* __restrict__ hist) {
    __shared__ __align__(32) _Float16 ebuf[2][16 * 256];   // 2 x 8 KB double buffer

    int tid  = threadIdx.x;
    int lane = tid & 31;
    int wave = tid >> 5;
    int rowBase = blockIdx.x * 128 + wave * 16;
    int l16 = lane & 15;
    int hi  = lane >> 4;        // 0: lanes 0-15, 1: lanes 16-31

    // --- load the 8 A fragments (16 rows x 256 dims, f16) ------------------
    // A layout (16-bit, 16x32), matching B: lanes 0-15 hold K 0..15,
    // lanes 16-31 hold K 16..31 of the same row.
    const _Float16* zrow = zt + (size_t)(rowBase + l16) * D;
    v16h A[8];
#pragma unroll
    for (int c = 0; c < 8; ++c) {
        int base = c * 32 + hi * 16;
        v8h p0 = *(const v8h*)(zrow + base);
        v8h p1 = *(const v8h*)(zrow + base + 8);
        v16h a;
#pragma unroll
        for (int i = 0; i < 8; ++i) { a[i] = p0[i]; a[i + 8] = p1[i]; }
        A[c] = a;
    }

    // --- prologue: async-stage codebook tile 0 into buffer 0 ---------------
    // Tile = 4096 halves = 512 x 16B chunks; each thread moves 2 chunks.
    async_b128(e16 + (size_t)tid * 8,         &ebuf[0][tid * 8]);
    async_b128(e16 + (size_t)(tid + 256) * 8, &ebuf[0][(tid + 256) * 8]);
    wait_async0();
    __syncthreads();

    float bestD[8];
    int   bestI[8];
#pragma unroll
    for (int r = 0; r < 8; ++r) { bestD[r] = 3.4e38f; bestI[r] = 0; }

    // --- sweep the 512 codes in 32 tiles of 16 columns ---------------------
    for (int kt = 0; kt < 32; ++kt) {
        int cur = kt & 1;

        // prefetch next tile into the other buffer (reads of it retired at
        // the barrier that ended iteration kt-1)
        if (kt + 1 < 32) {
            const _Float16* gsrc = e16 + (size_t)(kt + 1) * 16 * D;
            async_b128(gsrc + (size_t)tid * 8,         &ebuf[cur ^ 1][tid * 8]);
            async_b128(gsrc + (size_t)(tid + 256) * 8, &ebuf[cur ^ 1][(tid + 256) * 8]);
        }

        int col = kt * 16 + l16;                 // codebook row = B column
        // B layout (16-bit, 32x16): lane n holds column n; lanes 0-15 K=0..15,
        // lanes 16-31 K=16..31 -> 16 contiguous halves of the staged row.
        const _Float16* erow = &ebuf[cur][l16 * D + hi * 16];

        v8f acc = {};
#pragma unroll
        for (int c = 0; c < 8; ++c) {
            v16h Bf = *(const v16h*)(erow + c * 32);
            acc = __builtin_amdgcn_wmma_f32_16x16x32_f16(
                false, A[c], false, Bf, (short)0, acc, false, false);
        }

        float en = enorm[col];
#pragma unroll
        for (int r = 0; r < 8; ++r) {
            float sc = en - 2.0f * acc[r];
            if (sc < bestD[r]) { bestD[r] = sc; bestI[r] = col; }  // kt ascending -> first-min kept
        }

        // retire own prefetch, then publish LDS to all waves
        wait_async0();
        __syncthreads();
    }

    // --- cross-lane argmin within each 16-lane half (row-sharing lanes) ----
#pragma unroll
    for (int r = 0; r < 8; ++r) {
        float bd = bestD[r];
        int   bi = bestI[r];
#pragma unroll
        for (int m = 8; m >= 1; m >>= 1) {
            float od = __shfl_xor(bd, m, 32);
            int   oi = __shfl_xor(bi, m, 32);
            if (od < bd || (od == bd && oi < bi)) { bd = od; bi = oi; }
        }
        bestD[r] = bd; bestI[r] = bi;
    }

    // C/D layout: VGPR r holds row M = r + 8*hi. Lane 0 of each half writes.
    if (l16 == 0) {
#pragma unroll
        for (int r = 0; r < 8; ++r) {
            int row = rowBase + hi * 8 + r;
            int bi  = bestI[r];
            idx_i[row] = bi;
            idx_f[row] = (float)bi;
            atomicAdd(&hist[bi], 1);
        }
    }
}

// ---------------------------------------------------------------------------
// Gather z_q = e[idx] back to [B,D,T,h,w], accumulate exact f32 squared error.
// ---------------------------------------------------------------------------
__global__ __launch_bounds__(256) void vq_gather(const float* __restrict__ z,
                                                 const float* __restrict__ e,
                                                 const int* __restrict__ idx_i,
                                                 float* __restrict__ out,
                                                 float* __restrict__ loss) {
    __shared__ float red[256];
    long long o = (long long)blockIdx.x * 256 + threadIdx.x;   // o < ND
    int b   = (int)(o >> 22);               // D*S = 2^22
    int rem = (int)(o & ((1 << 22) - 1));
    int d   = rem >> 14;                    // S = 2^14
    int s   = rem & (S - 1);
    int n   = (b << 14) + s;

    int   k = idx_i[n];
    float q = e[k * D + d];
    float diff = q - z[o];
    out[o] = q;

    red[threadIdx.x] = diff * diff;
    __syncthreads();
    for (int st = 128; st > 0; st >>= 1) {
        if (threadIdx.x < st) red[threadIdx.x] += red[threadIdx.x + st];
        __syncthreads();
    }
    if (threadIdx.x == 0) atomicAdd(loss, red[0]);
}

// ---------------------------------------------------------------------------
// Finalize: perplexity from histogram, scaled loss.  One block of 512.
// ---------------------------------------------------------------------------
__global__ __launch_bounds__(512) void vq_final(const int* __restrict__ hist,
                                                const float* __restrict__ loss,
                                                float* __restrict__ out) {
    __shared__ float red[512];
    int t = threadIdx.x;
    float p = (float)hist[t] / (float)N;
    red[t] = p * logf(p + 1e-10f);
    __syncthreads();
    for (int st = 256; st > 0; st >>= 1) {
        if (t < st) red[t] += red[t + st];
        __syncthreads();
    }
    if (t == 0) {
        out[ND] = 1.25f * (*loss) / (float)ND;          // vq_loss (codebk + 0.25*commit)
        out[ND + 1 + N] = expf(-red[0]);                // perplexity
    }
}

// ---------------------------------------------------------------------------
extern "C" void kernel_launch(void* const* d_in, const int* in_sizes, int n_in,
                              void* d_out, int out_size, void* d_ws, size_t ws_size,
                              hipStream_t stream) {
    const float* z = (const float*)d_in[0];
    const float* e = (const float*)d_in[1];
    float* out = (float*)d_out;

    char* ws = (char*)d_ws;
    _Float16* zt    = (_Float16*)(ws);                       // N*D*2   = 67108864 B
    _Float16* e16   = (_Float16*)(ws + 67108864);            // K*D*2   = 262144 B
    float*    enorm = (float*)   (ws + 67371008);            // K*4     = 2048 B
    int*      idx_i = (int*)     (ws + 67373056);            // N*4     = 524288 B
    int*      hist  = (int*)     (ws + 67897344);            // K*4     = 2048 B
    float*    loss  = (float*)   (ws + 67899392);            // 4 B

    vq_zero     <<<2, 256, 0, stream>>>(hist, loss);
    vq_prep_e   <<<K, 256, 0, stream>>>(e, e16, enorm);
    vq_transpose<<<dim3(S / 32, D / 32, Bb), dim3(32, 8), 0, stream>>>(z, zt);
    vq_argmin   <<<N / 128, 256, 0, stream>>>(zt, e16, enorm, idx_i, out + ND + 1, hist);
    vq_gather   <<<(int)(ND / 256), 256, 0, stream>>>(z, e, idx_i, out, loss);
    vq_final    <<<1, 512, 0, stream>>>(hist, loss, out);
}
